// IterativeAttention_38903813767557
// MI455X (gfx1250) — compile-verified, matrix operand order checked by reading, not run on hardware
//
#include <hip/hip_runtime.h>

#define BATCH 16
#define ODIM  4096
#define NQ    512

typedef __attribute__((ext_vector_type(16))) _Float16 v16h;
typedef __attribute__((ext_vector_type(8)))  _Float16 v8h;
typedef __attribute__((ext_vector_type(8)))  float    v8f;

// ---------------------------------------------------------------------------
// WMMA helpers (CDNA5, wave32): D(16x16 f32) = A(16x32 f16) * B(32x16 f16) + C
// ---------------------------------------------------------------------------
__device__ __forceinline__ v8f wmma_f16(v16h a, v16h b, v8f c) {
  return __builtin_amdgcn_wmma_f32_16x16x32_f16(false, a, false, b, (short)0, c,
                                                false, false);
}

__device__ __forceinline__ v8f zero8() {
  v8f z;
#pragma unroll
  for (int i = 0; i < 8; i++) z[i] = 0.0f;
  return z;
}

// Load a 16x32 f16 fragment (valid as A from [M][K] row-major, or as B from
// [N][K] "contraction-contiguous" storage). Per ISA 16-bit A layout:
// lane L holds row (L&15); K = (L>>4)*8 + {0..7} in regs 0-3 and
// K = (L>>4)*8 + 16 + {0..7} in regs 4-7 -> two contiguous 16B loads.
__device__ __forceinline__ v16h load_frag16(const _Float16* base, int ld,
                                            int row0, int k0) {
  int lane = threadIdx.x & 31;
  int r  = row0 + (lane & 15);
  int kb = k0 + ((lane >> 4) << 3);
  const _Float16* p = base + (size_t)r * (size_t)ld + kb;
  v8h h0 = *(const v8h*)(p);
  v8h h1 = *(const v8h*)(p + 16);
  v16h a;
#pragma unroll
  for (int i = 0; i < 8; i++) { a[i] = h0[i]; a[8 + i] = h1[i]; }
  return a;
}

// ---------------------------------------------------------------------------
// Feature generators (positional embedding + 1-in linear), computed on the fly
// ---------------------------------------------------------------------------
__device__ __forceinline__ float feat_out(int k, int j, float xv,
                                          const float* wl, const float* bl) {
  if (k >= 63) return 0.0f;
  if (k < 42) return wl[k] * xv + bl[k];
  int t = k - 42;                       // 21-feature embed of c = j/64
  float c = (float)j * (1.0f / 64.0f);
  if (t == 0) return c;
  int q = t - 1;
  float f = (float)(1 << (q >> 1));
  float a = c * f;
  return (q & 1) ? __cosf(a) : __sinf(a);
}

__device__ __forceinline__ float feat_pos(int k, int pos, float xg,
                                          const float* wl, const float* bl,
                                          int kmax) {
  if (k >= kmax) return 0.0f;
  if (k < 42) return wl[k] * xg + bl[k];
  int t = k - 42;                       // interleaved 2-channel embed
  int part = t >> 1, ch = t & 1;
  float c = (float)(ch ? (pos & 63) : (pos >> 6)) * (1.0f / 64.0f);
  if (part == 0) return c;
  int q = part - 1;
  float f = (float)(1 << (q >> 1));
  float a = c * f;
  return (q & 1) ? __cosf(a) : __sinf(a);
}

// ---------------------------------------------------------------------------
// Utility kernels
// ---------------------------------------------------------------------------
__global__ void init_out_kernel(const float* __restrict__ x,
                                float* __restrict__ out) {
  int t = blockIdx.x * 256 + threadIdx.x;
  if (t >= BATCH * ODIM) return;
  int b = t >> 12, j = t & 4095;
  out[t] = x[b * 8192 + j];
}

__global__ void convert_pad_kernel(const float* __restrict__ W,
                                   _Float16* __restrict__ Wh,
                                   int o, int i, int ipad) {
  int t = blockIdx.x * 256 + threadIdx.x;
  if (t >= o * ipad) return;
  int r = t / ipad, c = t - r * ipad;
  Wh[t] = (_Float16)((c < i) ? W[r * i + c] : 0.0f);
}

// ---------------------------------------------------------------------------
// WMMA MLP trunk kernel. One wave per 16 positions.
// mode 0: out path (feat_out, K1=64)   -> eOut f16 [b][4096][64], vtOut f16 [b][64][4096]
// mode 1: pos path (feat_pos, K1=96)   -> same shapes
// mode 2: gathered at path (K1=64, 2 stages) -> eOut f16 [b][512][64] + eF32
// ---------------------------------------------------------------------------
__global__ __launch_bounds__(32) void mlp_kernel(
    int mode, int npos,
    const float* __restrict__ x, const int* __restrict__ idx,
    const float* __restrict__ wlin, const float* __restrict__ blin,
    const _Float16* __restrict__ W1h, const float* __restrict__ b1, int K1,
    const _Float16* __restrict__ W2h, const float* __restrict__ b2,
    const _Float16* __restrict__ W3h, const float* __restrict__ b3,
    _Float16* __restrict__ eOut, _Float16* __restrict__ vtOut,
    float* __restrict__ eF32) {
  __shared__ __align__(16) _Float16 ldsA[16 * 96];
  __shared__ __align__(16) _Float16 ldsH[16 * 64];
  int lane = threadIdx.x;
  int tiles_per_b = npos >> 4;
  int b = blockIdx.x / tiles_per_b;
  int posbase = (blockIdx.x % tiles_per_b) << 4;
  int ng = lane >> 4, nl = lane & 15;

  // Build 16 x K1 feature tile in LDS (A operand of stage 1)
  for (int t = lane; t < 16 * K1; t += 32) {
    int row = t / K1, k = t - row * K1;
    int p = posbase + row;
    float f;
    if (mode == 0) {
      f = feat_out(k, p, x[b * 8192 + p], wlin, blin);
    } else {
      int pos = (mode == 2) ? idx[b * NQ + p] : p;
      float xg = x[b * 8192 + ODIM + pos];
      f = feat_pos(k, pos, xg, wlin, blin, (mode == 2) ? 63 : 84);
    }
    ldsA[t] = (_Float16)f;
  }
  __syncthreads();

  // fc1 + relu
  v8f h[4];
#pragma unroll
  for (int nt = 0; nt < 4; nt++) {
    v8f acc = zero8();
    for (int kc = 0; kc < K1; kc += 32) {
      v16h a = load_frag16(ldsA, K1, 0, kc);
      v16h w = load_frag16(W1h, K1, nt * 16, kc);
      acc = wmma_f16(a, w, acc);
    }
    float bn = b1[nt * 16 + nl];
#pragma unroll
    for (int r = 0; r < 8; r++) acc[r] = fmaxf(acc[r] + bn, 0.0f);
    h[nt] = acc;
  }
#pragma unroll
  for (int nt = 0; nt < 4; nt++)
#pragma unroll
    for (int r = 0; r < 8; r++)
      ldsH[(r + 8 * ng) * 64 + nt * 16 + nl] = (_Float16)h[nt][r];
  __syncthreads();

  // fc2 (no relu on e / at_embed)
  v8f g[4];
#pragma unroll
  for (int nt = 0; nt < 4; nt++) {
    v8f acc = zero8();
#pragma unroll
    for (int kc = 0; kc < 64; kc += 32) {
      v16h a = load_frag16(ldsH, 64, 0, kc);
      v16h w = load_frag16(W2h, 64, nt * 16, kc);
      acc = wmma_f16(a, w, acc);
    }
    float bn = b2[nt * 16 + nl];
#pragma unroll
    for (int r = 0; r < 8; r++) acc[r] += bn;
    g[nt] = acc;
  }
#pragma unroll
  for (int nt = 0; nt < 4; nt++)
#pragma unroll
    for (int r = 0; r < 8; r++) {
      int p = posbase + r + 8 * ng;
      size_t ofs = ((size_t)b * npos + p) * 64 + nt * 16 + nl;
      eOut[ofs] = (_Float16)g[nt][r];
      if (eF32) eF32[ofs] = g[nt][r];
    }

  // fc3 on relu(e) -> value, stored TRANSPOSED [b][64][4096] so attention's
  // P@V B-fragments are contiguous 16B loads.
  if (W3h) {
    __syncthreads();
#pragma unroll
    for (int nt = 0; nt < 4; nt++)
#pragma unroll
      for (int r = 0; r < 8; r++)
        ldsH[(r + 8 * ng) * 64 + nt * 16 + nl] = (_Float16)fmaxf(g[nt][r], 0.0f);
    __syncthreads();
#pragma unroll
    for (int nt = 0; nt < 4; nt++) {
      v8f acc = zero8();
#pragma unroll
      for (int kc = 0; kc < 64; kc += 32) {
        v16h a = load_frag16(ldsH, 64, 0, kc);
        v16h w = load_frag16(W3h, 64, nt * 16, kc);
        acc = wmma_f16(a, w, acc);
      }
      float bn = b3[nt * 16 + nl];
#pragma unroll
      for (int r = 0; r < 8; r++) {
        int p = posbase + r + 8 * ng;
        vtOut[((size_t)b * 64 + nt * 16 + nl) * (size_t)ODIM + p] =
            (_Float16)(acc[r] + bn);
      }
    }
  }
}

// ---------------------------------------------------------------------------
// Flash attention: one wave per (batch, 16-query tile). Streams 4096 keys in
// 32-key chunks; online softmax; never materializes the 512x4096 logits.
// ---------------------------------------------------------------------------
__global__ __launch_bounds__(32) void attn_kernel(
    const _Float16* __restrict__ Q,   // [B][512][64]
    const _Float16* __restrict__ E,   // [B][4096][64] keys (row-major)
    const _Float16* __restrict__ VT,  // [B][64][4096]  values (transposed)
    float* __restrict__ O) {          // [B][512][64]
  __shared__ __align__(16) _Float16 ldsP[16 * 32];
  int lane = threadIdx.x;
  int b = blockIdx.x >> 5;
  int qt = blockIdx.x & 31;
  int ng = lane >> 4, nl = lane & 15;
  const _Float16* qb = Q + ((size_t)b * NQ + qt * 16) * 64;
  const _Float16* eb = E + (size_t)b * ODIM * 64;
  const _Float16* vb = VT + (size_t)b * 64 * ODIM;

  v16h qa0 = load_frag16(qb, 64, 0, 0);
  v16h qa1 = load_frag16(qb, 64, 0, 32);

  float m[8], l[8];
  v8f oAcc[4];
#pragma unroll
  for (int r = 0; r < 8; r++) { m[r] = -3.0e38f; l[r] = 0.0f; }
#pragma unroll
  for (int nt = 0; nt < 4; nt++) oAcc[nt] = zero8();

  for (int kb0 = 0; kb0 < ODIM; kb0 += 32) {
    // S = Q @ K^T for 32 keys (two 16x16 C tiles, K-dim 64 = 2 chunks)
    v8f s0 = zero8(), s1 = zero8();
    s0 = wmma_f16(qa0, load_frag16(eb + (size_t)kb0 * 64, 64, 0, 0), s0);
    s0 = wmma_f16(qa1, load_frag16(eb + (size_t)kb0 * 64, 64, 0, 32), s0);
    s1 = wmma_f16(qa0, load_frag16(eb + (size_t)(kb0 + 16) * 64, 64, 0, 0), s1);
    s1 = wmma_f16(qa1, load_frag16(eb + (size_t)(kb0 + 16) * 64, 64, 0, 32), s1);

    // Online softmax update per row (rows r+8*ng; N striped over 16 lanes)
#pragma unroll
    for (int r = 0; r < 8; r++) {
      float mx = fmaxf(s0[r], s1[r]);
#pragma unroll
      for (int d = 1; d < 16; d <<= 1) mx = fmaxf(mx, __shfl_xor(mx, d, 32));
      float mn = fmaxf(m[r], mx);
      float sc = __expf(m[r] - mn);
      float p0 = __expf(s0[r] - mn);
      float p1 = __expf(s1[r] - mn);
      float rs = p0 + p1;
#pragma unroll
      for (int d = 1; d < 16; d <<= 1) rs += __shfl_xor(rs, d, 32);
      l[r] = l[r] * sc + rs;
      m[r] = mn;
#pragma unroll
      for (int nt = 0; nt < 4; nt++) oAcc[nt][r] *= sc;
      int row = r + 8 * ng;
      ldsP[row * 32 + nl]      = (_Float16)p0;
      ldsP[row * 32 + 16 + nl] = (_Float16)p1;
    }
    __syncthreads();

    // O += P(16x32) @ V(32x64): B-fragments from transposed V, contiguous
    v16h pa = load_frag16(ldsP, 32, 0, 0);
#pragma unroll
    for (int nt = 0; nt < 4; nt++) {
      v16h vf = load_frag16(vb + (size_t)nt * 16 * ODIM + kb0, ODIM, 0, 0);
      oAcc[nt] = wmma_f16(pa, vf, oAcc[nt]);
    }
    __syncthreads();
  }

#pragma unroll
  for (int nt = 0; nt < 4; nt++)
#pragma unroll
    for (int r = 0; r < 8; r++) {
      int q = qt * 16 + r + 8 * ng;
      O[((size_t)b * NQ + q) * 64 + nt * 16 + nl] = oAcc[nt][r] / l[r];
    }
}

// ---------------------------------------------------------------------------
// Final MLP (192->64 relu, 64->64 relu, 64->1) + scatter into output at idx
// ---------------------------------------------------------------------------
__global__ __launch_bounds__(32) void final_kernel(
    const float* __restrict__ atF, const float* __restrict__ pAttn,
    const float* __restrict__ oAttn,
    const _Float16* __restrict__ W1h, const float* __restrict__ b1,
    const _Float16* __restrict__ W2h, const float* __restrict__ b2,
    const float* __restrict__ w3, const float* __restrict__ b3,
    const int* __restrict__ idx, float* __restrict__ out) {
  __shared__ __align__(16) _Float16 ldsF[16 * 192];
  __shared__ __align__(16) _Float16 ldsH[16 * 64];
  int lane = threadIdx.x;
  int b = blockIdx.x >> 5;
  int qt = blockIdx.x & 31;
  int ng = lane >> 4, nl = lane & 15;

  for (int t = lane; t < 16 * 192; t += 32) {
    int row = t / 192, k = t - row * 192;
    size_t base = ((size_t)b * NQ + qt * 16 + row) * 64;
    float f = (k < 64) ? atF[base + k]
              : (k < 128) ? pAttn[base + k - 64]
                          : oAttn[base + k - 128];
    ldsF[t] = (_Float16)f;
  }
  __syncthreads();

  // fc1 + relu
  v8f h[4];
#pragma unroll
  for (int nt = 0; nt < 4; nt++) {
    v8f acc = zero8();
#pragma unroll
    for (int kc = 0; kc < 192; kc += 32) {
      v16h a = load_frag16(ldsF, 192, 0, kc);
      v16h w = load_frag16(W1h, 192, nt * 16, kc);
      acc = wmma_f16(a, w, acc);
    }
    float bn = b1[nt * 16 + nl];
#pragma unroll
    for (int r = 0; r < 8; r++) acc[r] = fmaxf(acc[r] + bn, 0.0f);
    h[nt] = acc;
  }
#pragma unroll
  for (int nt = 0; nt < 4; nt++)
#pragma unroll
    for (int r = 0; r < 8; r++)
      ldsH[(r + 8 * ng) * 64 + nt * 16 + nl] = (_Float16)h[nt][r];
  __syncthreads();

  // fc2 + relu
  v8f g[4];
#pragma unroll
  for (int nt = 0; nt < 4; nt++) {
    v8f acc = zero8();
#pragma unroll
    for (int kc = 0; kc < 64; kc += 32) {
      v16h a = load_frag16(ldsH, 64, 0, kc);
      v16h w = load_frag16(W2h, 64, nt * 16, kc);
      acc = wmma_f16(a, w, acc);
    }
    float bn = b2[nt * 16 + nl];
#pragma unroll
    for (int r = 0; r < 8; r++) acc[r] = fmaxf(acc[r] + bn, 0.0f);
    g[nt] = acc;
  }

  // fc3: per-row dot with w3 (f32), reduce across 16 lanes, scatter
  float rv[8];
#pragma unroll
  for (int r = 0; r < 8; r++) {
    float s = 0.0f;
#pragma unroll
    for (int nt = 0; nt < 4; nt++) s += g[nt][r] * w3[nt * 16 + nl];
#pragma unroll
    for (int d = 1; d < 16; d <<= 1) s += __shfl_xor(s, d, 32);
    rv[r] = s + b3[0];
  }
  if (nl == 0) {
#pragma unroll
    for (int r = 0; r < 8; r++) {
      int q = qt * 16 + r + 8 * ng;
      out[b * ODIM + idx[b * NQ + q]] = rv[r];
    }
  }
}

// ---------------------------------------------------------------------------
// Host launcher
// ---------------------------------------------------------------------------
enum { L_INP_LIN, L_OUT_LIN, L_INP_FC1, L_INP_FC2, L_INP_FC3, L_OUT_FC1,
       L_OUT_FC2, L_OUT_FC3, L_AT_FC1, L_AT_FC2, L_FC1, L_FC2, L_FC3, NL };

extern "C" void kernel_launch(void* const* d_in, const int* in_sizes, int n_in,
                              void* d_out, int out_size, void* d_ws,
                              size_t ws_size, hipStream_t stream) {
  (void)n_in; (void)out_size; (void)ws_size;
  const float* x = nullptr;
  const int* idx = nullptr;
  const float* Wp[NL];
  const float* Bp[NL];

  if (in_sizes[0] == BATCH * 8192) {
    // insertion order: x, idx, then (w, b) per layer as written in reference
    x = (const float*)d_in[0];
    idx = (const int*)d_in[1];
    for (int i = 0; i < NL; i++) {
      Wp[i] = (const float*)d_in[2 + 2 * i];
      Bp[i] = (const float*)d_in[3 + 2 * i];
    }
  } else {
    // sorted pytree order: idx, {layers alphabetical, b before w}, x
    idx = (const int*)d_in[0];
    static const int s2i[NL] = {L_AT_FC1, L_AT_FC2, L_FC1, L_FC2, L_FC3,
                                L_INP_FC1, L_INP_FC2, L_INP_FC3, L_INP_LIN,
                                L_OUT_FC1, L_OUT_FC2, L_OUT_FC3, L_OUT_LIN};
    for (int s = 0; s < NL; s++) {
      Bp[s2i[s]] = (const float*)d_in[1 + 2 * s];
      Wp[s2i[s]] = (const float*)d_in[2 + 2 * s];
    }
    x = (const float*)d_in[1 + 2 * NL];
  }

  char* ws = (char*)d_ws;
  size_t off = 0;
  auto carve = [&](size_t bytes) -> char* {
    char* p = ws + off;
    off = (off + bytes + 255) & ~(size_t)255;
    return p;
  };

  _Float16* wh_inp_fc1 = (_Float16*)carve(64 * 96 * 2);
  _Float16* wh_inp_fc2 = (_Float16*)carve(64 * 64 * 2);
  _Float16* wh_inp_fc3 = (_Float16*)carve(64 * 64 * 2);
  _Float16* wh_out_fc1 = (_Float16*)carve(64 * 64 * 2);
  _Float16* wh_out_fc2 = (_Float16*)carve(64 * 64 * 2);
  _Float16* wh_out_fc3 = (_Float16*)carve(64 * 64 * 2);
  _Float16* wh_at_fc1  = (_Float16*)carve(64 * 64 * 2);
  _Float16* wh_at_fc2  = (_Float16*)carve(64 * 64 * 2);
  _Float16* wh_fc1     = (_Float16*)carve(64 * 192 * 2);
  _Float16* wh_fc2     = (_Float16*)carve(64 * 64 * 2);
  _Float16* e_out  = (_Float16*)carve((size_t)BATCH * ODIM * 64 * 2);
  _Float16* e_pos  = (_Float16*)carve((size_t)BATCH * ODIM * 64 * 2);
  _Float16* vt_out = (_Float16*)carve((size_t)BATCH * 64 * ODIM * 2);
  _Float16* vt_pos = (_Float16*)carve((size_t)BATCH * 64 * ODIM * 2);
  _Float16* q_at   = (_Float16*)carve((size_t)BATCH * NQ * 64 * 2);
  float* at_f32    = (float*)carve((size_t)BATCH * NQ * 64 * 4);
  float* attn_out  = (float*)carve((size_t)BATCH * NQ * 64 * 4);
  float* attn_pos  = (float*)carve((size_t)BATCH * NQ * 64 * 4);

  init_out_kernel<<<(BATCH * ODIM + 255) / 256, 256, 0, stream>>>(
      x, (float*)d_out);

  auto conv = [&](const float* W, _Float16* Wh, int o, int i, int ipad) {
    int n = o * ipad;
    convert_pad_kernel<<<(n + 255) / 256, 256, 0, stream>>>(W, Wh, o, i, ipad);
  };
  conv(Wp[L_INP_FC1], wh_inp_fc1, 64, 84, 96);
  conv(Wp[L_INP_FC2], wh_inp_fc2, 64, 64, 64);
  conv(Wp[L_INP_FC3], wh_inp_fc3, 64, 64, 64);
  conv(Wp[L_OUT_FC1], wh_out_fc1, 64, 63, 64);
  conv(Wp[L_OUT_FC2], wh_out_fc2, 64, 64, 64);
  conv(Wp[L_OUT_FC3], wh_out_fc3, 64, 64, 64);
  conv(Wp[L_AT_FC1],  wh_at_fc1,  64, 63, 64);
  conv(Wp[L_AT_FC2],  wh_at_fc2,  64, 64, 64);
  conv(Wp[L_FC1],     wh_fc1,     64, 192, 192);
  conv(Wp[L_FC2],     wh_fc2,     64, 64, 64);

  // Trunk MLPs (out path, pos path) and gathered query MLP
  mlp_kernel<<<BATCH * ODIM / 16, 32, 0, stream>>>(
      0, ODIM, x, idx, Wp[L_OUT_LIN], Bp[L_OUT_LIN],
      wh_out_fc1, Bp[L_OUT_FC1], 64, wh_out_fc2, Bp[L_OUT_FC2],
      wh_out_fc3, Bp[L_OUT_FC3], e_out, vt_out, (float*)nullptr);
  mlp_kernel<<<BATCH * ODIM / 16, 32, 0, stream>>>(
      1, ODIM, x, idx, Wp[L_INP_LIN], Bp[L_INP_LIN],
      wh_inp_fc1, Bp[L_INP_FC1], 96, wh_inp_fc2, Bp[L_INP_FC2],
      wh_inp_fc3, Bp[L_INP_FC3], e_pos, vt_pos, (float*)nullptr);
  mlp_kernel<<<BATCH * NQ / 16, 32, 0, stream>>>(
      2, NQ, x, idx, Wp[L_INP_LIN], Bp[L_INP_LIN],
      wh_at_fc1, Bp[L_AT_FC1], 64, wh_at_fc2, Bp[L_AT_FC2],
      (const _Float16*)nullptr, (const float*)nullptr, q_at,
      (_Float16*)nullptr, at_f32);

  // Flash attentions (out pair, pos pair)
  attn_kernel<<<BATCH * 32, 32, 0, stream>>>(q_at, e_out, vt_out, attn_out);
  attn_kernel<<<BATCH * 32, 32, 0, stream>>>(q_at, e_pos, vt_pos, attn_pos);

  // Final MLP + scatter
  final_kernel<<<BATCH * 32, 32, 0, stream>>>(
      at_f32, attn_pos, attn_out, wh_fc1, Bp[L_FC1], wh_fc2, Bp[L_FC2],
      Wp[L_FC3], Bp[L_FC3], idx, (float*)d_out);
}
